// RNNAttentionDecoder_18803366821933
// MI455X (gfx1250) — matrix operand order repaired and checked by reading, not compiled, not run on hardware
//
#include <hip/hip_runtime.h>
#include <hip/hip_bf16.h>

// ---------------------------------------------------------------------------
// RNN attention decoder for MI455X (gfx1250, wave32, WMMA).
//   * Prepass: weights fp32 [K,N] -> bf16 [N,K] (transposed) in d_ws, so B
//     fragments are two contiguous 16B loads per lane. All weights (~23 MB)
//     stay resident in the 192 MB L2 across the 48-step scan.
//   * Activations are produced in bf16 by their producer kernels, so the
//     WMMA GEMM reads A fragments directly from global memory (row-major
//     bf16, 16 contiguous k per lane, same k order as B) -- no LDS, no barriers,
//     no conversions in the GEMM hot loop.
//   * GEMM: 256 threads = 8 waves; each wave = 1 M-tile x 2 N-tiles sharing
//     one A fragment -> two back-to-back v_wmma_f32_16x16x32_bf16 per chunk.
//   * GRU gates / softmax attention are tiny -> fp32 VALU kernels.
// ---------------------------------------------------------------------------

typedef __attribute__((ext_vector_type(16))) __bf16 v16bf;
typedef __attribute__((ext_vector_type(8)))  __bf16 v8bf;
typedef __attribute__((ext_vector_type(8)))  float  v8f;
typedef __attribute__((ext_vector_type(4)))  float  v4f;

#define T_STEPS 48
#define BATCH   32
#define SLEN    128
#define EDIM    512
#define HDIM    512
#define VOCAB   16000
#define NSPECS  4

// fp32 -> bf16, round-to-nearest-even (bit trick; only used off the hot path)
__device__ __forceinline__ __bf16 f2bf(float f) {
    union { float f; unsigned u; } uf; uf.f = f;
    unsigned r = uf.u + 0x7FFFu + ((uf.u >> 16) & 1u);
    union { unsigned short s; __bf16 b; } ub;
    ub.s = (unsigned short)(r >> 16);
    return ub.b;
}

// ---------------------------------------------------------------------------
// Prepass: W fp32 [K,N] (row-major)  ->  Wt bf16 [N,K] (row-major, transposed)
// ---------------------------------------------------------------------------
__global__ void k_cvt_transpose(const float* __restrict__ W,
                                __bf16* __restrict__ Wt, int K, int N) {
    long long idx = (long long)blockIdx.x * blockDim.x + threadIdx.x;
    long long total = (long long)K * N;
    if (idx >= total) return;
    int k = (int)(idx / N);
    int n = (int)(idx % N);           // coalesced read side
    Wt[(long long)n * K + k] = f2bf(W[idx]);
}

// ---------------------------------------------------------------------------
// WMMA GEMM: OUT[32,N] = act( A[32,K](bf16) @ Wt^T (bf16 [N,K]) + bias )
// gridDim.x = N/128. 8 waves: wave w -> mtile (w&1), N-tile pair (w>>1).
// Each wave: one A fragment feeds two accumulators (2 adjacent N tiles).
// ---------------------------------------------------------------------------
__global__ __launch_bounds__(256, 4)
void k_gemm_wmma(const __bf16* __restrict__ A, int K,
                 const __bf16* __restrict__ Wt,
                 const float* __restrict__ bias,
                 float* __restrict__ OUT, int ldo, int act,
                 __bf16* __restrict__ OUT_bf) {
    const int tid  = threadIdx.x;
    const int wave = tid >> 5;
    const int lane = tid & 31;
    const int mt   = wave & 1;                    // M rows 0-15 / 16-31
    const int nt0  = blockIdx.x * 8 + (wave >> 1) * 2;
    const int col0 = nt0 * 16 + (lane & 15);
    const int col1 = col0 + 16;
    const int ksel = lane >> 4;                   // lanes 16-31: upper K half
    const int arow = mt * 16 + (lane & 15);

    // A frag: lanes 0-15 hold K 0-15; lanes 16-31 hold K 16-31 (contiguous).
    const __bf16* ap  = A  + (long long)arow * K + ksel * 16;
    // B frag: lanes 0-15 hold K 0-15; lanes 16-31 hold K 16-31 (contiguous).
    const __bf16* bp0 = Wt + (long long)col0 * K + ksel * 16;
    const __bf16* bp1 = Wt + (long long)col1 * K + ksel * 16;

    v8f acc0 = {0.f, 0.f, 0.f, 0.f, 0.f, 0.f, 0.f, 0.f};
    v8f acc1 = {0.f, 0.f, 0.f, 0.f, 0.f, 0.f, 0.f, 0.f};

#pragma unroll 2
    for (int kc = 0; kc < K; kc += 32) {
        v8bf alo = *(const v8bf*)(ap + kc);
        v8bf ahi = *(const v8bf*)(ap + kc + 8);
        v16bf af;
#pragma unroll
        for (int i = 0; i < 8; ++i) { af[i] = alo[i]; af[8 + i] = ahi[i]; }

        v8bf b0l = *(const v8bf*)(bp0 + kc);
        v8bf b0h = *(const v8bf*)(bp0 + kc + 8);
        v16bf bf0;
#pragma unroll
        for (int i = 0; i < 8; ++i) { bf0[i] = b0l[i]; bf0[8 + i] = b0h[i]; }

        v8bf b1l = *(const v8bf*)(bp1 + kc);
        v8bf b1h = *(const v8bf*)(bp1 + kc + 8);
        v16bf bf1;
#pragma unroll
        for (int i = 0; i < 8; ++i) { bf1[i] = b1l[i]; bf1[8 + i] = b1h[i]; }

        if (kc + 32 < K) {                         // prefetch next K chunk of B
            __builtin_prefetch(bp0 + kc + 32, 0, 3);
            __builtin_prefetch(bp1 + kc + 32, 0, 3);
        }

        acc0 = __builtin_amdgcn_wmma_f32_16x16x32_bf16(
                   false, af, false, bf0, (short)0, acc0, false, false);
        acc1 = __builtin_amdgcn_wmma_f32_16x16x32_bf16(
                   false, af, false, bf1, (short)0, acc1, false, false);
    }

    // C/D layout: vgpr r -> row r + 8*(lane>>4); col = lane&15
    const float bv0 = bias ? bias[col0] : 0.f;
    const float bv1 = bias ? bias[col1] : 0.f;
#pragma unroll
    for (int r = 0; r < 8; ++r) {
        int row = mt * 16 + (lane >> 4) * 8 + r;
        float v0 = acc0[r] + bv0;
        float v1 = acc1[r] + bv1;
        if (act) { v0 = tanhf(v0); v1 = tanhf(v1); }
        OUT[(long long)row * ldo + col0] = v0;
        OUT[(long long)row * ldo + col1] = v1;
        if (OUT_bf) {
            OUT_bf[(long long)row * ldo + col0] = f2bf(v0);
            OUT_bf[(long long)row * ldo + col1] = f2bf(v1);
        }
    }
}

// ---------------------------------------------------------------------------
// Embedding + input feeding: x_bf[b] = [ word_emb + special_emb , o_prev ]
// ---------------------------------------------------------------------------
__global__ void k_embed(const int* __restrict__ ids_t,
                        const float* __restrict__ wemb,
                        const float* __restrict__ semb,
                        const __bf16* __restrict__ o_bf,
                        __bf16* __restrict__ x_bf) {
    int idx = blockIdx.x * blockDim.x + threadIdx.x;   // 32*512
    int b = idx >> 9, e = idx & 511;
    int id  = ids_t[b];
    int wid = id - NSPECS; if (wid < 0) wid = 0;
    int sid = id < NSPECS ? id : NSPECS;
    float v = wemb[(long long)wid * EDIM + e] + semb[sid * EDIM + e];
    x_bf[b * 1024 + e]       = f2bf(v);
    x_bf[b * 1024 + 512 + e] = o_bf[b * HDIM + e];
}

// ---------------------------------------------------------------------------
// GRU gates: h_new from gx, gh; writes h (fp32), h_bf, and xo_bf[:,512:]
// ---------------------------------------------------------------------------
__global__ void k_gru(const float* __restrict__ gx, const float* __restrict__ gh,
                      float* __restrict__ h, __bf16* __restrict__ h_bf,
                      __bf16* __restrict__ xo_bf) {
    int idx = blockIdx.x * blockDim.x + threadIdx.x;   // 32*512
    int b = idx >> 9, j = idx & 511;
    const float* gxb = gx + b * 1536;
    const float* ghb = gh + b * 1536;
    float r  = 1.f / (1.f + __expf(-(gxb[j]        + ghb[j])));
    float z  = 1.f / (1.f + __expf(-(gxb[512 + j]  + ghb[512 + j])));
    float n  = tanhf(gxb[1024 + j] + r * ghb[1024 + j]);
    float hn = (1.f - z) * n + z * h[idx];
    h[idx] = hn;
    h_bf[idx] = f2bf(hn);
    xo_bf[b * 1024 + 512 + j] = f2bf(hn);
}

// ---------------------------------------------------------------------------
// Attention (one block per batch row): scores -> softmax -> context vector
// Writes bf16 context vector into xo_bf[:,0:512].
// ---------------------------------------------------------------------------
__global__ __launch_bounds__(128)
void k_attn(const float* __restrict__ q, const float* __restrict__ ctx,
            const int* __restrict__ cmask, __bf16* __restrict__ xo_bf) {
    __shared__ float qs[HDIM];
    __shared__ float sc[SLEN];
    __shared__ float red[SLEN];
    const int b = blockIdx.x, s = threadIdx.x;

    for (int i = s; i < HDIM; i += 128) qs[i] = q[b * HDIM + i];
    __syncthreads();

    const float* crow = ctx + ((long long)b * SLEN + s) * HDIM;
    float acc = 0.f;
    for (int k = 0; k < HDIM; k += 4) {
        v4f cv = *(const v4f*)(crow + k);
        acc += cv[0] * qs[k] + cv[1] * qs[k + 1] + cv[2] * qs[k + 2] + cv[3] * qs[k + 3];
    }
    if (cmask[b * SLEN + s] == 0) acc = -1e9f;
    sc[s] = acc; red[s] = acc;
    __syncthreads();
    for (int off = 64; off > 0; off >>= 1) {
        if (s < off) red[s] = fmaxf(red[s], red[s + off]);
        __syncthreads();
    }
    float m = red[0];
    __syncthreads();
    float e = __expf(sc[s] - m);
    sc[s] = e; red[s] = e;
    __syncthreads();
    for (int off = 64; off > 0; off >>= 1) {
        if (s < off) red[s] += red[s + off];
        __syncthreads();
    }
    float inv = 1.f / red[0];

    // c[h] for h = s*4 .. s*4+3 (coalesced across the block)
    v4f cacc = {0.f, 0.f, 0.f, 0.f};
    const float* cb = ctx + (long long)b * SLEN * HDIM + s * 4;
    for (int ss = 0; ss < SLEN; ++ss) {
        float a = sc[ss] * inv;
        v4f cv = *(const v4f*)(cb + (long long)ss * HDIM);
        cacc[0] += a * cv[0]; cacc[1] += a * cv[1];
        cacc[2] += a * cv[2]; cacc[3] += a * cv[3];
    }
    __bf16* dst = xo_bf + b * 1024 + s * 4;
    dst[0] = f2bf(cacc[0]); dst[1] = f2bf(cacc[1]);
    dst[2] = f2bf(cacc[2]); dst[3] = f2bf(cacc[3]);
}

// ---------------------------------------------------------------------------
// State init / finalize
// ---------------------------------------------------------------------------
__global__ void k_init(const float* __restrict__ hidden,
                       const float* __restrict__ prev,
                       float* __restrict__ h, __bf16* __restrict__ h_bf,
                       __bf16* __restrict__ o_bf) {
    int idx = blockIdx.x * blockDim.x + threadIdx.x;   // 32*512
    float hv = hidden[idx];
    h[idx]    = hv;
    h_bf[idx] = f2bf(hv);
    o_bf[idx] = f2bf(prev[idx]);
}

__global__ void k_final(const float* __restrict__ h, const float* __restrict__ o,
                        float* __restrict__ tail) {
    int idx = blockIdx.x * blockDim.x + threadIdx.x;   // 32*512
    tail[idx]                = h[idx];
    tail[BATCH * HDIM + idx] = o[idx];
}

// ---------------------------------------------------------------------------
// Host driver
// ---------------------------------------------------------------------------
extern "C" void kernel_launch(void* const* d_in, const int* in_sizes, int n_in,
                              void* d_out, int out_size, void* d_ws, size_t ws_size,
                              hipStream_t stream) {
    const int*   ids      = (const int*)d_in[0];
    // d_in[1] = lengths (unused by reference math)
    const float* word_emb = (const float*)d_in[2];
    const float* spec_emb = (const float*)d_in[3];
    const float* hidden   = (const float*)d_in[4];
    const float* context  = (const float*)d_in[5];
    const int*   cmask    = (const int*)d_in[6];
    const float* prev_out = (const float*)d_in[7];
    const float* Wx = (const float*)d_in[8];   // [1024,1536]
    const float* Wh = (const float*)d_in[9];   // [512,1536]
    const float* bx = (const float*)d_in[10];  // [1536]
    const float* bh = (const float*)d_in[11];  // [1536]
    const float* Wa = (const float*)d_in[12];  // [512,512]
    const float* Wo = (const float*)d_in[13];  // [1024,512]
    const float* Wg = (const float*)d_in[14];  // [512,16000]
    const float* bg = (const float*)d_in[15];  // [16000]
    float* out = (float*)d_out;

    // --- workspace layout (256B aligned) ---
    char* ws = (char*)d_ws;
    size_t off = 0;
    auto take = [&](size_t bytes) -> char* {
        char* p = ws + off;
        off = (off + bytes + 255) & ~(size_t)255;
        return p;
    };
    __bf16* Wt_x  = (__bf16*)take((size_t)1536 * 1024 * 2);
    __bf16* Wt_h  = (__bf16*)take((size_t)1536 * 512  * 2);
    __bf16* Wt_a  = (__bf16*)take((size_t)512  * 512  * 2);
    __bf16* Wt_o  = (__bf16*)take((size_t)512  * 1024 * 2);
    __bf16* Wt_g  = (__bf16*)take((size_t)VOCAB * 512 * 2);
    __bf16* x_bf  = (__bf16*)take((size_t)BATCH * 1024 * 2);  // [e_t | o_prev]
    __bf16* h_bf  = (__bf16*)take((size_t)BATCH * HDIM * 2);
    __bf16* xo_bf = (__bf16*)take((size_t)BATCH * 1024 * 2);  // [c | h_new]
    __bf16* o_bf  = (__bf16*)take((size_t)BATCH * HDIM * 2);
    float*  gx    = (float*)take((size_t)BATCH * 1536 * 4);
    float*  gh    = (float*)take((size_t)BATCH * 1536 * 4);
    float*  hbuf  = (float*)take((size_t)BATCH * HDIM * 4);
    float*  qbuf  = (float*)take((size_t)BATCH * HDIM * 4);
    float*  obuf  = (float*)take((size_t)BATCH * HDIM * 4);
    (void)ws_size; (void)in_sizes; (void)n_in; (void)out_size;

    // --- prepass: weight transpose + bf16 convert (L2-resident afterwards) ---
    auto cvt = [&](const float* W, __bf16* Wt, int K, int N) {
        long long total = (long long)K * N;
        int blocks = (int)((total + 255) / 256);
        k_cvt_transpose<<<blocks, 256, 0, stream>>>(W, Wt, K, N);
    };
    cvt(Wx, Wt_x, 1024, 1536);
    cvt(Wh, Wt_h, 512,  1536);
    cvt(Wa, Wt_a, 512,  512);
    cvt(Wo, Wt_o, 1024, 512);
    cvt(Wg, Wt_g, 512,  VOCAB);

    k_init<<<64, 256, 0, stream>>>(hidden, prev_out, hbuf, h_bf, o_bf);

    // --- sequential scan over time ---
    for (int t = 0; t < T_STEPS; ++t) {
        k_embed<<<64, 256, 0, stream>>>(ids + t * BATCH, word_emb, spec_emb,
                                        o_bf, x_bf);
        // gx = x @ Wx + bx   [32,1024]x[1024,1536]
        k_gemm_wmma<<<12, 256, 0, stream>>>(x_bf, 1024, Wt_x, bx, gx, 1536, 0,
                                            nullptr);
        // gh = h @ Wh + bh   [32,512]x[512,1536]
        k_gemm_wmma<<<12, 256, 0, stream>>>(h_bf, 512, Wt_h, bh, gh, 1536, 0,
                                            nullptr);
        // GRU gates -> h_new (fp32 + bf16) and xo_bf[:,512:]
        k_gru<<<64, 256, 0, stream>>>(gx, gh, hbuf, h_bf, xo_bf);
        // q = h_new @ Wa     [32,512]x[512,512]
        k_gemm_wmma<<<4, 256, 0, stream>>>(h_bf, 512, Wt_a, nullptr, qbuf, 512,
                                           0, nullptr);
        // attention -> c (bf16) into xo_bf[:,0:512]
        k_attn<<<BATCH, 128, 0, stream>>>(qbuf, context, cmask, xo_bf);
        // o = tanh([c,h] @ Wo)   [32,1024]x[1024,512]  (fp32 + bf16 outputs)
        k_gemm_wmma<<<4, 256, 0, stream>>>(xo_bf, 1024, Wt_o, nullptr, obuf,
                                           512, 1, o_bf);
        // logits = o @ Wg + bg   [32,512]x[512,16000] -> scores[t]
        k_gemm_wmma<<<125, 256, 0, stream>>>(o_bf, 512, Wt_g, bg,
                                             out + (long long)t * BATCH * VOCAB,
                                             VOCAB, 0, nullptr);
    }

    // h_fin, o_fin appended after scores
    k_final<<<64, 256, 0, stream>>>(hbuf, obuf,
                                    out + (long long)T_STEPS * BATCH * VOCAB);
}
